// First_Decoder_48318382080489
// MI455X (gfx1250) — matrix operand order repaired and checked by reading, not compiled, not run on hardware
//
#include <hip/hip_runtime.h>

typedef __attribute__((ext_vector_type(16))) _Float16 v16h;
typedef __attribute__((ext_vector_type(8)))  float    v8f;

union AFrag { v16h v; _Float16 h[16]; };
union H4 { _Float16 h[4]; unsigned long long u; };

__device__ __forceinline__ v8f wmma16(v16h a, v16h b, v8f c) {
  // D = A(16x32 f16) * B(32x16 f16) + C(16x16 f32)
  return __builtin_amdgcn_wmma_f32_16x16x32_f16(false, a, false, b, (short)0, c,
                                                false, false);
}

// A-matrix 16x32 f16 per-lane element->K mapping used here (contiguous order; B operands use the same):
// lane: m = lane&15, hf = lane>>4; element e (0..15): k = hf*16 + e
__device__ __forceinline__ int kpat(int e, int hf) {
  return hf * 16 + e;
}

#define SEQ 200
#define DM  512
#define NH  8
#define DH  64
#define DFF 2048
#define BSZ 128
#define ROWS (BSZ * SEQ)   // 25600

// ---------------------------------------------------------------------------
// Weight prep: W[K][N] f32 -> Wt[N][K] f16 (B-operand layout), 4 k's per thread
// ---------------------------------------------------------------------------
__global__ __launch_bounds__(256) void prep_wt(const float* __restrict__ W,
                                               _Float16* __restrict__ Wt,
                                               int K, int N, int total4) {
  int i4 = blockIdx.x * 256 + threadIdx.x;
  if (i4 >= total4) return;
  int idx = i4 * 4;
  int n = idx / K, k = idx - n * K;
  H4 t;
#pragma unroll
  for (int j = 0; j < 4; ++j) t.h[j] = (_Float16)W[(size_t)(k + j) * N + n];
  *reinterpret_cast<unsigned long long*>(Wt + idx) = t.u;
}

// ---------------------------------------------------------------------------
// h = emb[x] + positional encoding (even/odd pair shares one angle)
// ---------------------------------------------------------------------------
__global__ __launch_bounds__(256) void embed_pos(const int* __restrict__ x,
                                                 const float* __restrict__ emb,
                                                 float* __restrict__ H,
                                                 int total2) {
  int i2 = blockIdx.x * 256 + threadIdx.x;
  if (i2 >= total2) return;
  int dp = i2 & (DM / 2 - 1);       // pair index within row
  int bs = i2 >> 8;                 // DM/2 = 256 = 2^8
  int s  = bs % SEQ;
  int tok = x[bs];
  int d0 = dp * 2;
  // 10000^(d0/512) = exp(ln(10000) * d0 / 512)
  float cst = __expf((float)d0 * (9.210340371976184f / (float)DM));
  float ang = (float)s / cst;
  float sn, cs;
  __sincosf(ang, &sn, &cs);
  const float2 ev = *reinterpret_cast<const float2*>(emb + (size_t)tok * DM + d0);
  float2 o; o.x = ev.x + sn; o.y = ev.y + cs;
  *reinterpret_cast<float2*>(H + (size_t)i2 * 2) = o;
}

// ---------------------------------------------------------------------------
// Generic WMMA GEMM: C[M,N] = act(A[M,K] @ Wt' + bias)
//   A: f32 row-major, Wt: f16 [N][K], C: f32 row-major
//   grid = (M/32, N/512), block = 256 (8 waves)
//   Requirements: M%32==0, N%512==0, K%64==0
// ---------------------------------------------------------------------------
__global__ __launch_bounds__(256) void gemm_wmma(const float* __restrict__ A,
                                                 const _Float16* __restrict__ Wt,
                                                 const float* __restrict__ bias,
                                                 float* __restrict__ C,
                                                 int M, int N, int K, int relu) {
  __shared__ __align__(32) _Float16 ldsA[32 * 64];
  int tid  = threadIdx.x;
  int wave = tid >> 5, lane = tid & 31;
  int hf = lane >> 4, ln = lane & 15;
  int m0 = blockIdx.x * 32;
  int n0 = blockIdx.y * 512;
  int mt = wave & 1;     // which 16-row tile
  int ng = wave >> 1;    // n-tile group 0..3

  v8f acc[8];
  v8f zero = {0.f, 0.f, 0.f, 0.f, 0.f, 0.f, 0.f, 0.f};
#pragma unroll
  for (int j = 0; j < 8; ++j) acc[j] = zero;

  for (int kb = 0; kb < K; kb += 64) {
    // cooperative A panel load: 32 rows x 64 k, f32 -> f16, float4 granularity
#pragma unroll
    for (int j = 0; j < 2; ++j) {
      int i4 = tid + j * 256;           // 512 float4s = 2048 elements
      int r = i4 >> 4, c = (i4 & 15) * 4;
      const float4 av = *reinterpret_cast<const float4*>(
          A + (size_t)(m0 + r) * K + kb + c);
      H4 t;
      t.h[0] = (_Float16)av.x; t.h[1] = (_Float16)av.y;
      t.h[2] = (_Float16)av.z; t.h[3] = (_Float16)av.w;
      *reinterpret_cast<unsigned long long*>(&ldsA[r * 64 + c]) = t.u;
    }
    __syncthreads();
#pragma unroll
    for (int kc = 0; kc < 2; ++kc) {
      AFrag a;
      int arow = mt * 16 + ln;
#pragma unroll
      for (int e = 0; e < 16; ++e) a.h[e] = ldsA[arow * 64 + kc * 32 + kpat(e, hf)];
#pragma unroll
      for (int j = 0; j < 8; ++j) {
        int n = n0 + (ng + 4 * j) * 16 + ln;
        const v16h* bp = reinterpret_cast<const v16h*>(
            Wt + (size_t)n * K + kb + kc * 32 + hf * 16);
        acc[j] = wmma16(a.v, *bp, acc[j]);
      }
    }
    __syncthreads();
  }
  // epilogue: C layout -> m = r + 8*hf, n = lane&15
#pragma unroll
  for (int j = 0; j < 8; ++j) {
    int n = n0 + (ng + 4 * j) * 16 + ln;
    float bv = bias ? bias[n] : 0.0f;
#pragma unroll
    for (int r = 0; r < 8; ++r) {
      int m = m0 + mt * 16 + hf * 8 + r;
      float v = acc[j][r] + bv;
      if (relu) v = fmaxf(v, 0.0f);
      C[(size_t)m * N + n] = v;
    }
  }
}

// ---------------------------------------------------------------------------
// Flash attention, one wave per (query-block of 16, head, batch)
//   Q,K,V: [B, S, 512] f32 (head h occupies columns h*64..h*64+63)
//   grid = (ceil(S/16)=13, H=8, B=128), block = 32
// ---------------------------------------------------------------------------
__global__ __launch_bounds__(32) void attn_wmma(const float* __restrict__ Q,
                                                const float* __restrict__ Kb,
                                                const float* __restrict__ Vb,
                                                float* __restrict__ O,
                                                int causal) {
  __shared__ __align__(32) _Float16 Plds[16 * 32];
  __shared__ __align__(32) _Float16 Vlds[64 * 32];
  int lane = threadIdx.x;
  int hf = lane >> 4, ln = lane & 15;
  int qb = blockIdx.x, h = blockIdx.y, b = blockIdx.z;
  int q0 = qb * 16;

  // Q fragments (A-operand): row = q0 + ln (clamped), two K=32 chunks over dh=64
  AFrag aq[2];
  {
    int qr = q0 + ln; if (qr > SEQ - 1) qr = SEQ - 1;
    const float* qp = Q + (size_t)(b * SEQ + qr) * DM + h * DH;
#pragma unroll
    for (int c = 0; c < 2; ++c)
#pragma unroll
      for (int e = 0; e < 16; ++e)
        aq[c].h[e] = (_Float16)qp[c * 32 + kpat(e, hf)];
  }

  v8f zero = {0.f, 0.f, 0.f, 0.f, 0.f, 0.f, 0.f, 0.f};
  v8f o[4];
#pragma unroll
  for (int j = 0; j < 4; ++j) o[j] = zero;
  float rm[8], rl[8];
#pragma unroll
  for (int r = 0; r < 8; ++r) { rm[r] = -1e30f; rl[r] = 0.f; }

  int nkb = (SEQ + 31) / 32;                       // 7 key blocks (padded to 224)
  if (causal) { int mx = (q0 + 15) >> 5; if (mx + 1 < nkb) nkb = mx + 1; }

  for (int kb = 0; kb < nkb; ++kb) {
    int k0 = kb * 32;
    // ---- scores: T = Q @ K^T (two 16-key tiles) ----
    v8f t0 = zero, t1 = zero;
#pragma unroll
    for (int nt = 0; nt < 2; ++nt) {
      int kr = k0 + nt * 16 + ln; if (kr > SEQ - 1) kr = SEQ - 1;
      const float* kp = Kb + (size_t)(b * SEQ + kr) * DM + h * DH;
#pragma unroll
      for (int c = 0; c < 2; ++c) {
        AFrag bk;                                  // B-operand: n=key, e->d consecutive
        const float* src = kp + c * 32 + hf * 16;
#pragma unroll
        for (int e = 0; e < 16; ++e) bk.h[e] = (_Float16)src[e];
        if (nt == 0) t0 = wmma16(aq[c].v, bk.v, t0);
        else         t1 = wmma16(aq[c].v, bk.v, t1);
      }
    }
    // ---- scale + mask ----
#pragma unroll
    for (int r = 0; r < 8; ++r) {
      int qg = q0 + hf * 8 + r;
      int key0 = k0 + ln, key1 = k0 + 16 + ln;
      float s0 = t0[r] * 0.125f;                   // 1/sqrt(64)
      float s1 = t1[r] * 0.125f;
      if (key0 >= SEQ || (causal && key0 > qg)) s0 = -1e30f;
      if (key1 >= SEQ || (causal && key1 > qg)) s1 = -1e30f;
      t0[r] = s0; t1[r] = s1;
    }
    // ---- online softmax update ----
#pragma unroll
    for (int r = 0; r < 8; ++r) {
      float vmax = fmaxf(t0[r], t1[r]);
#pragma unroll
      for (int m = 1; m <= 8; m <<= 1) vmax = fmaxf(vmax, __shfl_xor(vmax, m, 32));
      float mn = fmaxf(rm[r], vmax);
      float corr = __expf(rm[r] - mn);
      rm[r] = mn;
      float p0 = __expf(t0[r] - mn);
      float p1 = __expf(t1[r] - mn);
      float rs = p0 + p1;
#pragma unroll
      for (int m = 1; m <= 8; m <<= 1) rs += __shfl_xor(rs, m, 32);
      rl[r] = rl[r] * corr + rs;
#pragma unroll
      for (int j = 0; j < 4; ++j) o[j][r] *= corr;
      Plds[(hf * 8 + r) * 32 + ln]      = (_Float16)p0;
      Plds[(hf * 8 + r) * 32 + 16 + ln] = (_Float16)p1;
    }
    // ---- V block -> LDS transposed [d][k] f16 ----
    {
      int kr = k0 + lane;
      bool valid = kr < SEQ;
      int krc = valid ? kr : SEQ - 1;
      const float* vp = Vb + (size_t)(b * SEQ + krc) * DM + h * DH;
#pragma unroll
      for (int d = 0; d < 64; ++d)
        Vlds[d * 32 + lane] = valid ? (_Float16)vp[d] : (_Float16)0.f;
    }
    // ---- O += P @ V ----
    AFrag ap;
#pragma unroll
    for (int e = 0; e < 16; ++e) ap.h[e] = Plds[ln * 32 + kpat(e, hf)];
#pragma unroll
    for (int dt = 0; dt < 4; ++dt) {
      const v16h* bv = reinterpret_cast<const v16h*>(&Vlds[(dt * 16 + ln) * 32 + hf * 16]);
      o[dt] = wmma16(ap.v, *bv, o[dt]);
    }
  }
  // ---- epilogue: O /= rowsum, store ----
#pragma unroll
  for (int r = 0; r < 8; ++r) {
    float inv = 1.0f / rl[r];
    int q = q0 + hf * 8 + r;
    if (q < SEQ) {
#pragma unroll
      for (int dt = 0; dt < 4; ++dt)
        O[(size_t)(b * SEQ + q) * DM + h * DH + dt * 16 + ln] = o[dt][r] * inv;
    }
  }
}

// ---------------------------------------------------------------------------
// Out[row] = LayerNorm(X[row] + Delta[row]) * g + be   (one block per row)
// ---------------------------------------------------------------------------
__global__ __launch_bounds__(256) void residual_ln(const float* __restrict__ X,
                                                   const float* __restrict__ Dl,
                                                   const float* __restrict__ g,
                                                   const float* __restrict__ be,
                                                   float* __restrict__ Out) {
  __shared__ float red[256];
  __shared__ float mean_s, var_s;
  int tid = threadIdx.x;
  size_t base = (size_t)blockIdx.x * DM;
  const float2 xv = *reinterpret_cast<const float2*>(X + base + 2 * tid);
  const float2 dv = *reinterpret_cast<const float2*>(Dl + base + 2 * tid);
  float v0 = xv.x + dv.x;
  float v1 = xv.y + dv.y;
  red[tid] = v0 + v1;
  __syncthreads();
  for (int s = 128; s > 0; s >>= 1) {
    if (tid < s) red[tid] += red[tid + s];
    __syncthreads();
  }
  if (tid == 0) mean_s = red[0] * (1.0f / DM);
  __syncthreads();
  float m = mean_s;
  float d0 = v0 - m, d1 = v1 - m;
  red[tid] = d0 * d0 + d1 * d1;
  __syncthreads();
  for (int s = 128; s > 0; s >>= 1) {
    if (tid < s) red[tid] += red[tid + s];
    __syncthreads();
  }
  if (tid == 0) var_s = red[0] * (1.0f / DM);
  __syncthreads();
  float inv = rsqrtf(var_s + 1e-5f);
  const float2 gv = *reinterpret_cast<const float2*>(g + 2 * tid);
  const float2 bv = *reinterpret_cast<const float2*>(be + 2 * tid);
  float2 ov;
  ov.x = d0 * inv * gv.x + bv.x;
  ov.y = d1 * inv * gv.y + bv.y;
  *reinterpret_cast<float2*>(Out + base + 2 * tid) = ov;
}

// ---------------------------------------------------------------------------
extern "C" void kernel_launch(void* const* d_in, const int* in_sizes, int n_in,
                              void* d_out, int out_size, void* d_ws, size_t ws_size,
                              hipStream_t stream) {
  (void)in_sizes; (void)n_in; (void)out_size; (void)ws_size;
  const int*   x    = (const int*)d_in[0];
  const float* enc  = (const float*)d_in[1];
  const float* emb  = (const float*)d_in[2];
  // d_in[3..10]: Wq_m, Wk_m, Wv_m, Wp_m, Wq_c, Wk_c, Wv_c, Wp_c
  const float* lnw  = (const float*)d_in[11];
  const float* lnb  = (const float*)d_in[12];
  const float* W1   = (const float*)d_in[13];
  const float* b1   = (const float*)d_in[14];
  const float* W2   = (const float*)d_in[15];
  const float* b2   = (const float*)d_in[16];
  float* out = (float*)d_out;

  const size_t BUF = (size_t)ROWS * DM;            // 13.1M floats
  float* buf0 = (float*)d_ws;
  float* buf1 = buf0 + BUF;                        // also start of [ROWS,DFF] region
  float* buf2 = buf1 + BUF;
  float* buf3 = buf2 + BUF;
  float* buf4 = buf3 + BUF;
  float* buf5 = buf4 + BUF;
  _Float16* wt = (_Float16*)(buf5 + BUF);          // 8 x [512][512] f16
  _Float16* w1t = wt + 8 * (size_t)DM * DM;        // [2048][512] f16
  _Float16* w2t = w1t + (size_t)DFF * DM;          // [512][2048] f16

  // ---- weight prep ----
  for (int i = 0; i < 8; ++i) {
    prep_wt<<<(DM * DM / 4 + 255) / 256, 256, 0, stream>>>(
        (const float*)d_in[3 + i], wt + (size_t)i * DM * DM, DM, DM, DM * DM / 4);
  }
  prep_wt<<<(DM * DFF / 4 + 255) / 256, 256, 0, stream>>>(W1, w1t, DM, DFF,
                                                          DM * DFF / 4);
  prep_wt<<<(DFF * DM / 4 + 255) / 256, 256, 0, stream>>>(W2, w2t, DFF, DM,
                                                          DFF * DM / 4);

  // ---- embedding + positional ----
  embed_pos<<<(int)(BUF / 2 / 256), 256, 0, stream>>>(x, emb, buf0, (int)(BUF / 2));

  dim3 g512(ROWS / 32, 1), g2048(ROWS / 32, DFF / 512);
  dim3 gat(13, NH, BSZ);
  _Float16* Wq_m = wt + 0 * (size_t)DM * DM;
  _Float16* Wk_m = wt + 1 * (size_t)DM * DM;
  _Float16* Wv_m = wt + 2 * (size_t)DM * DM;
  _Float16* Wp_m = wt + 3 * (size_t)DM * DM;
  _Float16* Wq_c = wt + 4 * (size_t)DM * DM;
  _Float16* Wk_c = wt + 5 * (size_t)DM * DM;
  _Float16* Wv_c = wt + 6 * (size_t)DM * DM;
  _Float16* Wp_c = wt + 7 * (size_t)DM * DM;

  // ---- masked self-attention ----
  gemm_wmma<<<g512, 256, 0, stream>>>(buf0, Wq_m, nullptr, buf1, ROWS, DM, DM, 0);
  gemm_wmma<<<g512, 256, 0, stream>>>(buf0, Wk_m, nullptr, buf2, ROWS, DM, DM, 0);
  gemm_wmma<<<g512, 256, 0, stream>>>(buf0, Wv_m, nullptr, buf3, ROWS, DM, DM, 0);
  attn_wmma<<<gat, 32, 0, stream>>>(buf1, buf2, buf3, buf4, 1);
  gemm_wmma<<<g512, 256, 0, stream>>>(buf4, Wp_m, nullptr, buf5, ROWS, DM, DM, 0);
  residual_ln<<<ROWS, 256, 0, stream>>>(buf0, buf5, lnw, lnb, buf0);

  // ---- cross-attention ----
  gemm_wmma<<<g512, 256, 0, stream>>>(buf0, Wq_c, nullptr, buf1, ROWS, DM, DM, 0);
  gemm_wmma<<<g512, 256, 0, stream>>>(enc,  Wk_c, nullptr, buf2, ROWS, DM, DM, 0);
  gemm_wmma<<<g512, 256, 0, stream>>>(enc,  Wv_c, nullptr, buf3, ROWS, DM, DM, 0);
  attn_wmma<<<gat, 32, 0, stream>>>(buf1, buf2, buf3, buf4, 0);
  gemm_wmma<<<g512, 256, 0, stream>>>(buf4, Wp_c, nullptr, buf5, ROWS, DM, DM, 0);
  residual_ln<<<ROWS, 256, 0, stream>>>(buf0, buf5, lnw, lnb, buf0);

  // ---- FFN (intermediate [ROWS,DFF] spans buf1..buf4) ----
  gemm_wmma<<<g2048, 256, 0, stream>>>(buf0, w1t, b1, buf1, ROWS, DFF, DM, 1);
  gemm_wmma<<<g512, 256, 0, stream>>>(buf1, w2t, b2, buf5, ROWS, DM, DFF, 0);
  residual_ln<<<ROWS, 256, 0, stream>>>(buf0, buf5, lnw, lnb, out);
}
